// Anchor2TestsFuser_86328842650387
// MI455X (gfx1250) — matrix operand order repaired and checked by reading, not compiled, not run on hardware
//
#include <hip/hip_runtime.h>

// ---------------------------------------------------------------------------
// CDNA5 (gfx1250, wave32) fused attention for Anchor2TestsFuser.
//
// Math refactor (exact):
//   scores[a,h,t] = (u[a,h,:] . tf[t,:,a] + qkb[a,h]) / 8
//     u[:,h,:]    = G_h @ af,   G_h = Kw_h^T Qw_h   (bf16 WMMA GEMMs)
//   prob = softmax_t(scores)                (output #2)
//   m[a,h,:]      = sum_t prob[a,h,t] * tf[t,:,a]
//   out[a,:]      = sum_h W_h @ m[a,h] + b_fused     (bf16 WMMA GEMM)
//     W_h[o,i]    = sum_e fc_w[o,e*4+h] * v_w[e*4+h,i]
//     b_fused[o]  = fc_b[o] + sum_c fc_w[o,c]*v_b[c]   (uses sum_t prob == 1)
// Single HBM pass over test_feats (134MB) dominates: ~5.8us @ 23.3TB/s.
// Tile staging uses GLOBAL_LOAD_ASYNC_TO_LDS_B128 + s_wait_asynccnt when the
// toolchain exposes the builtins (guarded by __has_builtin); otherwise falls
// back to a register-double-buffered global->VGPR->LDS path.
// ---------------------------------------------------------------------------

typedef __attribute__((ext_vector_type(16))) __bf16 v16bf;
typedef __attribute__((ext_vector_type(8)))  float  v8f;
typedef __attribute__((ext_vector_type(4)))  float  v4f;
typedef __attribute__((ext_vector_type(4)))  unsigned int v4u;
typedef __attribute__((ext_vector_type(4)))  int v4i;

#if defined(__has_builtin)
#if __has_builtin(__builtin_amdgcn_global_load_async_to_lds_b128) && \
    __has_builtin(__builtin_amdgcn_s_wait_asynccnt)
#define HAVE_ASYNC_LDS 1
#endif
#endif

#ifdef HAVE_ASYNC_LDS
typedef __attribute__((address_space(1))) v4i gv4i;  // global 16B vector
typedef __attribute__((address_space(3))) v4i lv4i;  // LDS 16B vector

// Issue 64B (4 x b128) async copy of one tile row: global -> LDS.
__device__ inline void tile_issue_async(const float* gsrc, float* ldst) {
  gv4i* g = (gv4i*)gsrc;
  lv4i* l = (lv4i*)ldst;
  __builtin_amdgcn_global_load_async_to_lds_b128(g, l, 0, 0);
  __builtin_amdgcn_global_load_async_to_lds_b128(g, l, 16, 0);
  __builtin_amdgcn_global_load_async_to_lds_b128(g, l, 32, 0);
  __builtin_amdgcn_global_load_async_to_lds_b128(g, l, 48, 0);
}
#endif

union FragBF {
  v16bf v;
  v4u   q[2];
};

// A fragment, 16x32 bf16, A stored row-major [M][K] (ld = K-stride).
// Per ISA 7.12.2: lanes 0-15 rows M=0-15 hold K 0..7 (v0..3) and 16..23 (v4..7);
// lanes 16-31 hold K 8..15 and 24..31.
__device__ inline v16bf load_a_frag(const __bf16* base, int ld, int lane, int row0, int k0) {
  FragBF f;
  const __bf16* p = base + (size_t)(row0 + (lane & 15)) * ld + (k0 + ((lane >> 4) << 3));
  f.q[0] = *(const v4u*)(p);
  f.q[1] = *(const v4u*)(p + 16);
  return f.v;
}

// B fragment, 32x16 bf16, B stored col-major as [N][K] (ld = K-stride).
// Per ISA: lane column n = lane&15; lanes 0-15 hold K 0..7 and 16..23, lanes 16-31 K 8..15 and 24..31.
__device__ inline v16bf load_b_frag(const __bf16* base, int ld, int lane, int col0, int k0) {
  FragBF f;
  const __bf16* p = base + (size_t)(col0 + (lane & 15)) * ld + (k0 + ((lane >> 4) << 3));
  f.q[0] = *(const v4u*)(p);
  f.q[1] = *(const v4u*)(p + 16);
  return f.v;
}

// ---------------------------------------------------------------------------
// K1: layout conversion + small fused vectors (VALU).
// ---------------------------------------------------------------------------
__global__ __launch_bounds__(256) void prep_kernel(
    const float* __restrict__ af,  const float* __restrict__ qw,
    const float* __restrict__ qb,  const float* __restrict__ kw,
    const float* __restrict__ kb,  const float* __restrict__ vw,
    const float* __restrict__ vb,  const float* __restrict__ fcw,
    const float* __restrict__ fcb,
    __bf16* __restrict__ afT,  __bf16* __restrict__ QwT,
    __bf16* __restrict__ KwT,  __bf16* __restrict__ Fc,
    __bf16* __restrict__ Vc,   float* __restrict__ g,
    float* __restrict__ chv,   float* __restrict__ cst,
    float* __restrict__ bfu)
{
  const int tid = blockIdx.x * blockDim.x + threadIdx.x;
  const int nth = gridDim.x * blockDim.x;

  // afT[a][e] = af[e,a]  (B operand, col-major [N=a][K=e])
  for (int idx = tid; idx < 2048 * 256; idx += nth) {
    int a = idx >> 8, e = idx & 255;
    afT[idx] = (__bf16)af[e * 2048 + a];
  }
  // QwT[h][e][d] = qw[d*4+h, e] ; KwT[h][i][d] = kw[d*4+h, i]
  for (int idx = tid; idx < 4 * 256 * 64; idx += nth) {
    int d = idx & 63, e = (idx >> 6) & 255, h = idx >> 14;
    QwT[idx] = (__bf16)qw[(d * 4 + h) * 256 + e];
    KwT[idx] = (__bf16)kw[(d * 4 + h) * 256 + e];
  }
  // Fc[h][o][e] = fc_w[o, e*4+h] ; Vc[h][i][e] = v_w[e*4+h, i]
  for (int idx = tid; idx < 4 * 256 * 256; idx += nth) {
    int e = idx & 255, oi = (idx >> 8) & 255, h = idx >> 16;
    Fc[idx] = (__bf16)fcw[oi * 1024 + e * 4 + h];
    Vc[idx] = (__bf16)vw[(e * 4 + h) * 256 + oi];
  }
  // g[h][i] = sum_d kw[d*4+h,i]*qb[d*4+h]; chv[h][e] = sum_d qw[d*4+h,e]*kb[d*4+h]
  for (int idx = tid; idx < 4 * 256; idx += nth) {
    int e = idx & 255, h = idx >> 8;
    float sg = 0.f, sc2 = 0.f;
    for (int d = 0; d < 64; ++d) {
      sg  += kw[(d * 4 + h) * 256 + e] * qb[d * 4 + h];
      sc2 += qw[(d * 4 + h) * 256 + e] * kb[d * 4 + h];
    }
    g[idx] = sg; chv[idx] = sc2;
  }
  // cst[h] = sum_d qb[d*4+h]*kb[d*4+h]
  for (int idx = tid; idx < 4; idx += nth) {
    float s = 0.f;
    for (int d = 0; d < 64; ++d) s += qb[d * 4 + idx] * kb[d * 4 + idx];
    cst[idx] = s;
  }
  // bfu[o] = fc_b[o] + sum_c fc_w[o,c]*v_b[c]
  for (int idx = tid; idx < 256; idx += nth) {
    float s = fcb[idx];
    for (int c = 0; c < 1024; ++c) s += fcw[idx * 1024 + c] * vb[c];
    bfu[idx] = s;
  }
}

// ---------------------------------------------------------------------------
// K1b: qkb[a,h] = chv[h,:] . af[:,a] + cst[h]
// ---------------------------------------------------------------------------
__global__ __launch_bounds__(256) void qkb_kernel(
    const float* __restrict__ af, const float* __restrict__ chv,
    const float* __restrict__ cst, float* __restrict__ qkbo)
{
  int idx = blockIdx.x * blockDim.x + threadIdx.x;   // a*4 + h, 8192 total
  if (idx >= 2048 * 4) return;
  int a = idx >> 2, h = idx & 3;
  float s = cst[h];
  for (int e = 0; e < 256; ++e) s += chv[h * 256 + e] * af[e * 2048 + a];
  qkbo[idx] = s;
}

// ---------------------------------------------------------------------------
// K2: generic bf16 GEMM, C(bf16) = A(row-major MxK) @ B(col-major [N][K]).
// One wave per 16x16 tile; v_wmma_f32_16x16x32_bf16 K-loop.
// ---------------------------------------------------------------------------
__global__ __launch_bounds__(32) void gemm_bf16(
    const __bf16* __restrict__ A, int sAz, int lda,
    const __bf16* __restrict__ B, int sBz, int ldb,
    __bf16* __restrict__ C, int sCz, int ldc, int K)
{
  const int lane = threadIdx.x;
  const int n0 = blockIdx.x * 16, m0 = blockIdx.y * 16, z = blockIdx.z;
  const __bf16* Az = A + (size_t)z * sAz;
  const __bf16* Bz = B + (size_t)z * sBz;
  __bf16* Cz = C + (size_t)z * sCz;

  v8f acc = {};
  for (int k0 = 0; k0 < K; k0 += 32) {
    v16bf av = load_a_frag(Az, lda, lane, m0, k0);
    v16bf bv = load_b_frag(Bz, ldb, lane, n0, k0);
    acc = __builtin_amdgcn_wmma_f32_16x16x32_bf16(false, av, false, bv,
                                                  (short)0, acc, false, false);
  }
  const int nl = n0 + (lane & 15);
  const int rb = m0 + ((lane >> 4) << 3);
#pragma unroll
  for (int j = 0; j < 8; ++j)
    Cz[(size_t)(rb + j) * ldc + nl] = (__bf16)acc[j];
}

// ---------------------------------------------------------------------------
// K3: U_h(256 x 2048) = G_h @ afT^T + g_h, scattered to u[a][h][i] (f32).
// ---------------------------------------------------------------------------
__global__ __launch_bounds__(32) void gemm_u_kernel(
    const __bf16* __restrict__ G, const __bf16* __restrict__ afT,
    const float* __restrict__ g, float* __restrict__ u)
{
  const int lane = threadIdx.x;
  const int a0 = blockIdx.x * 16, i0 = blockIdx.y * 16, h = blockIdx.z;
  const __bf16* Az = G + (size_t)h * 65536;   // [i][e], lda=256

  v8f acc = {};
  for (int k0 = 0; k0 < 256; k0 += 32) {
    v16bf av = load_a_frag(Az, 256, lane, i0, k0);
    v16bf bv = load_b_frag(afT, 256, lane, a0, k0);
    acc = __builtin_amdgcn_wmma_f32_16x16x32_bf16(false, av, false, bv,
                                                  (short)0, acc, false, false);
  }
  const int a  = a0 + (lane & 15);
  const int ib = i0 + ((lane >> 4) << 3);
#pragma unroll
  for (int j = 0; j < 8; ++j)
    u[((size_t)a * 4 + h) * 256 + ib + j] = acc[j] + g[h * 256 + ib + j];
}

// ---------------------------------------------------------------------------
// K4: main attention kernel. 128 blocks x 256 threads; 16 anchors per block.
// Pass 1: scores (VALU, quad-split dots + shfl_xor reduce), softmax -> prob.
// Pass 2: m accumulation (register accumulators, L2-resident re-read).
// Final:  out tile = Wcat(256x1024) @ m(1024x16) via WMMA, + b_fused.
// Dynamic LDS layout (116,224 B):
//   u_s  bf16 [64][264]  @ 0      (pad 264 => conflict-free quad reads)
//   Xs0  f32  [256][16]  @ 33792  (tile buffer A)
//   Xs1  f32  [256][16]  @ 50176  (tile buffer B, async double-buffer)
//   sc   f32  [64][65]   @ 66560
//   m_sb bf16 [16][1032] @ 83200  (pad 1032 => spread WMMA B-frag banks)
// ---------------------------------------------------------------------------
__global__ __launch_bounds__(256) void attn_kernel(
    const float* __restrict__ tf, const float* __restrict__ u,
    const float* __restrict__ qkb, const __bf16* __restrict__ Wcat,
    const float* __restrict__ bfu, float* __restrict__ out,
    float* __restrict__ prob_out)
{
  extern __shared__ char smem[];
  __bf16* u_s  = (__bf16*)(smem);            // 64*264*2 = 33792 B
  float*  Xs0  = (float*)(smem + 33792);     // 256*16*4 = 16384 B
  float*  Xs1  = (float*)(smem + 50176);     // 256*16*4 = 16384 B
  float*  sc   = (float*)(smem + 66560);     // 64*65*4  = 16640 B
  __bf16* m_sb = (__bf16*)(smem + 83200);    // 16*1032*2= 33024 B

  const int tid = threadIdx.x;
  const int a0  = blockIdx.x * 16;

  // stage u (f32 global -> bf16 LDS), padded row stride 264
  for (int idx = tid; idx < 16 * 1024; idx += 256) {
    int a = idx >> 10, rest = idx & 1023;
    u_s[((a << 2) | (rest >> 8)) * 264 + (rest & 255)] =
        (__bf16)u[(size_t)(a0 + a) * 1024 + rest];
  }

  const int combo = tid >> 2;          // local a*4 + h
  const int split = tid & 3;           // quad split of the i-dimension
  const int ca = combo >> 2;
  const int hh = combo & 3;
  const float myqkb = qkb[(size_t)(a0 + ca) * 4 + hh];
  const float* tfb = tf + a0;
  const __bf16* up = u_s + combo * 264;

  // ---------------- pass 1: scores ----------------
#ifdef HAVE_ASYNC_LDS
  tile_issue_async(tfb + (size_t)tid * 2048, Xs0 + tid * 16);
  __builtin_amdgcn_s_wait_asynccnt(0);
  __syncthreads();   // u_s staged + tile 0 landed
  for (int t = 0; t < 64; ++t) {
    float* Xc = (t & 1) ? Xs1 : Xs0;
    float* Xn = (t & 1) ? Xs0 : Xs1;
    if (t + 1 < 64)   // async-prefetch next tile into the other buffer
      tile_issue_async(tfb + (size_t)(t + 1) * 524288 + (size_t)tid * 2048,
                       Xn + tid * 16);
    float acc = 0.f;
#pragma unroll
    for (int ii = 0; ii < 64; ++ii) {
      int i = (ii << 2) + split;       // interleaved split -> distinct LDS banks
      acc += (float)up[i] * Xc[i * 16 + ca];
    }
    acc += __shfl_xor(acc, 1);
    acc += __shfl_xor(acc, 2);
    if (split == 0) sc[combo * 65 + t] = (acc + myqkb) * 0.125f;  // /sqrt(64)
    __builtin_amdgcn_s_wait_asynccnt(0);
    __syncthreads();
  }
#else
  v4f r0, r1, r2, r3;
  {
    const v4f* src = (const v4f*)(tfb + (size_t)tid * 2048);
    r0 = src[0]; r1 = src[1]; r2 = src[2]; r3 = src[3];
  }
  __syncthreads();   // u_s staged
  for (int t = 0; t < 64; ++t) {
    v4f* dst = (v4f*)(Xs0 + tid * 16);
    dst[0] = r0; dst[1] = r1; dst[2] = r2; dst[3] = r3;
    __syncthreads();
    if (t + 1 < 64) {   // register double-buffer: issue next-tile loads now
      const v4f* src = (const v4f*)(tfb + (size_t)(t + 1) * 524288 + (size_t)tid * 2048);
      r0 = src[0]; r1 = src[1]; r2 = src[2]; r3 = src[3];
    }
    float acc = 0.f;
#pragma unroll
    for (int ii = 0; ii < 64; ++ii) {
      int i = (ii << 2) + split;
      acc += (float)up[i] * Xs0[i * 16 + ca];
    }
    acc += __shfl_xor(acc, 1);
    acc += __shfl_xor(acc, 2);
    if (split == 0) sc[combo * 65 + t] = (acc + myqkb) * 0.125f;
    __syncthreads();
  }
#endif

  // ---------------- softmax over T (+ prob output) ----------------
  if (tid < 64) {
    float* row = sc + tid * 65;
    float mx = -1e30f;
#pragma unroll
    for (int t = 0; t < 64; ++t) mx = fmaxf(mx, row[t]);
    float s = 0.f;
#pragma unroll
    for (int t = 0; t < 64; ++t) { float e = __expf(row[t] - mx); row[t] = e; s += e; }
    float inv = 1.f / s;
    float* pd = prob_out + ((size_t)(a0 + (tid >> 2)) * 4 + (tid & 3)) * 64;
#pragma unroll
    for (int t = 0; t < 64; ++t) { float p = row[t] * inv; row[t] = p; pd[t] = p; }
  }
  __syncthreads();

  // ---------------- pass 2: m[a,h,:] = sum_t prob * tf (L2 re-read) --------
  float macc[64];
#pragma unroll
  for (int ii = 0; ii < 64; ++ii) macc[ii] = 0.f;

#ifdef HAVE_ASYNC_LDS
  tile_issue_async(tfb + (size_t)tid * 2048, Xs0 + tid * 16);
  __builtin_amdgcn_s_wait_asynccnt(0);
  __syncthreads();
  for (int t = 0; t < 64; ++t) {
    float* Xc = (t & 1) ? Xs1 : Xs0;
    float* Xn = (t & 1) ? Xs0 : Xs1;
    if (t + 1 < 64)
      tile_issue_async(tfb + (size_t)(t + 1) * 524288 + (size_t)tid * 2048,
                       Xn + tid * 16);
    float p = sc[combo * 65 + t];
#pragma unroll
    for (int ii = 0; ii < 64; ++ii) {
      int i = (ii << 2) + split;
      macc[ii] += p * Xc[i * 16 + ca];
    }
    __builtin_amdgcn_s_wait_asynccnt(0);
    __syncthreads();
  }
#else
  {
    const v4f* src = (const v4f*)(tfb + (size_t)tid * 2048);
    r0 = src[0]; r1 = src[1]; r2 = src[2]; r3 = src[3];
  }
  for (int t = 0; t < 64; ++t) {
    v4f* dst = (v4f*)(Xs0 + tid * 16);
    dst[0] = r0; dst[1] = r1; dst[2] = r2; dst[3] = r3;
    __syncthreads();
    if (t + 1 < 64) {
      const v4f* src = (const v4f*)(tfb + (size_t)(t + 1) * 524288 + (size_t)tid * 2048);
      r0 = src[0]; r1 = src[1]; r2 = src[2]; r3 = src[3];
    }
    float p = sc[combo * 65 + t];
#pragma unroll
    for (int ii = 0; ii < 64; ++ii) {
      int i = (ii << 2) + split;
      macc[ii] += p * Xs0[i * 16 + ca];
    }
    __syncthreads();
  }
#endif

  // write m as bf16 B-operand (col-major [a][c], c = h*256+i, padded 1032)
#pragma unroll
  for (int ii = 0; ii < 64; ++ii) {
    int i = (ii << 2) + split;
    m_sb[ca * 1032 + (hh << 8) + i] = (__bf16)macc[ii];
  }
  __syncthreads();

  // ---------------- final: out tile = Wcat @ m + b_fused (WMMA) ----------
  const int wave = tid >> 5, lane = tid & 31;
#pragma unroll
  for (int mt = 0; mt < 2; ++mt) {
    const int o0 = wave * 32 + mt * 16;
    v8f acc8 = {};
    for (int k0 = 0; k0 < 1024; k0 += 32) {
      v16bf av = load_a_frag(Wcat, 1024, lane, o0, k0);   // global (L2-resident)
      v16bf bv = load_b_frag(m_sb, 1032, lane, 0, k0);    // LDS
      acc8 = __builtin_amdgcn_wmma_f32_16x16x32_bf16(false, av, false, bv,
                                                     (short)0, acc8, false, false);
    }
    const int aa = a0 + (lane & 15);
    const int ob = o0 + ((lane >> 4) << 3);
    float* dst = out + (size_t)aa * 256 + ob;
#pragma unroll
    for (int j = 0; j < 8; ++j) dst[j] = acc8[j] + bfu[ob + j];
  }
}

// ---------------------------------------------------------------------------
// Host launcher.
// ---------------------------------------------------------------------------
extern "C" void kernel_launch(void* const* d_in, const int* in_sizes, int n_in,
                              void* d_out, int out_size, void* d_ws, size_t ws_size,
                              hipStream_t stream) {
  (void)in_sizes; (void)n_in; (void)out_size; (void)ws_size;

  const float* af  = (const float*)d_in[0];
  const float* tf  = (const float*)d_in[1];
  const float* qw  = (const float*)d_in[2];
  const float* qb  = (const float*)d_in[3];
  const float* kw  = (const float*)d_in[4];
  const float* kb  = (const float*)d_in[5];
  const float* vw  = (const float*)d_in[6];
  const float* vb  = (const float*)d_in[7];
  const float* fcw = (const float*)d_in[8];
  const float* fcb = (const float*)d_in[9];

  char* ws = (char*)d_ws;
  size_t off = 0;
  auto carve = [&](size_t bytes) -> char* {
    char* p = ws + off;
    off += (bytes + 255) & ~(size_t)255;
    return p;
  };
  __bf16* afT  = (__bf16*)carve((size_t)2048 * 256 * 2);   // 1 MB
  __bf16* QwT  = (__bf16*)carve((size_t)4 * 256 * 64 * 2);
  __bf16* KwT  = (__bf16*)carve((size_t)4 * 256 * 64 * 2);
  __bf16* Fc   = (__bf16*)carve((size_t)4 * 256 * 256 * 2);
  __bf16* Vc   = (__bf16*)carve((size_t)4 * 256 * 256 * 2);
  __bf16* G    = (__bf16*)carve((size_t)4 * 256 * 256 * 2);
  __bf16* Wcat = (__bf16*)carve((size_t)256 * 1024 * 2);
  float*  u    = (float*) carve((size_t)2048 * 4 * 256 * 4); // 8 MB
  float*  qkbp = (float*) carve((size_t)2048 * 4 * 4);
  float*  g    = (float*) carve((size_t)4 * 256 * 4);
  float*  chv  = (float*) carve((size_t)4 * 256 * 4);
  float*  cst  = (float*) carve((size_t)4 * 4);
  float*  bfu  = (float*) carve((size_t)256 * 4);
  // total ~11.8 MB of workspace

  prep_kernel<<<512, 256, 0, stream>>>(af, qw, qb, kw, kb, vw, vb, fcw, fcb,
                                       afT, QwT, KwT, Fc, Vc, g, chv, cst, bfu);
  qkb_kernel<<<32, 256, 0, stream>>>(af, chv, cst, qkbp);

  // G_h = Kw_h^T @ Qw_h    (M=256 i, N=256 e, K=64 d)
  gemm_bf16<<<dim3(16, 16, 4), 32, 0, stream>>>(KwT, 256 * 64, 64,
                                                QwT, 256 * 64, 64,
                                                G, 256 * 256, 256, 64);
  // Wcat[:, h*256 + i] = Fc_h @ Vc_h  (M=256 o, N=256 i, K=256 e)
  gemm_bf16<<<dim3(16, 16, 4), 32, 0, stream>>>(Fc, 256 * 256, 256,
                                                Vc, 256 * 256, 256,
                                                Wcat, 256, 1024, 256);
  // u[a][h][i] = (G_h @ af)[i,a] + g_h[i]
  gemm_u_kernel<<<dim3(128, 16, 4), 32, 0, stream>>>(G, afT, g, u);

  float* outp  = (float*)d_out;          // (A=2048, E=256)
  float* probp = outp + 2048 * 256;      // (A, H=4, 1, T=64)
  attn_kernel<<<128, 256, 116224, stream>>>(tf, u, qkbp, Wcat, bfu, outp, probp);
}
